// MORNPatient_54709293416894
// MI455X (gfx1250) — matrix operand order repaired and checked by reading, not compiled, not run on hardware
//
#include <hip/hip_runtime.h>
#include <math.h>

#define NGENE 50000
#define NHID 128
#define EGG 400000
#define NPATCH 4096
#define WSIDIM 1024
#define INV_SQRT_DK 0.17677669529663688f

typedef __attribute__((ext_vector_type(16))) __bf16 v16bf;
typedef __attribute__((ext_vector_type(8))) float v8f;

union BFrag { v16bf v; unsigned short us[16]; uint4 q[2]; };

__device__ __forceinline__ unsigned short f2bf(float f) {
  unsigned u = __float_as_uint(f);
  u += 0x7FFFu + ((u >> 16) & 1u);
  return (unsigned short)(u >> 16);
}
__device__ __forceinline__ unsigned fkey(float f) {   // order-preserving uint encoding
  unsigned u = __float_as_uint(f);
  return (u & 0x80000000u) ? ~u : (u | 0x80000000u);
}
__device__ __forceinline__ float fdec(unsigned k) {
  return __uint_as_float((k & 0x80000000u) ? (k ^ 0x80000000u) : ~k);
}
__device__ __forceinline__ float gelu_exact(float x) {
  return 0.5f * x * (1.0f + erff(x * 0.70710678118654752f));
}

// ---------------- weight packing into WMMA B-fragment order (bf16) ----------------
// slot t = ((kc*8 + ntile)*32 + lane); element i of lane's v16bf holds
// B[k = kc*32 + 16*sel + i][col = ntile*16 + (lane&15)], sel = lane>>4.
__global__ void k_pack_w(const float* __restrict__ W, unsigned short* __restrict__ P) {
  int t = blockIdx.x * blockDim.x + threadIdx.x;
  if (t >= 1024) return;
  const int kc = t >> 8, nt = (t >> 5) & 7, lane = t & 31;
  const int col = nt * 16 + (lane & 15), sel = lane >> 4;
  unsigned short* dst = P + t * 16;
#pragma unroll
  for (int i = 0; i < 16; ++i) {
    const int k = kc * 32 + 16 * sel + i;
    dst[i] = f2bf(W[k * 128 + col]);
  }
}

// block-diagonal pack of per-head [4][32][32] relation matrix into 128x128
__global__ void k_pack_bd(const float* __restrict__ R, unsigned short* __restrict__ P) {
  int t = blockIdx.x * blockDim.x + threadIdx.x;
  if (t >= 1024) return;
  const int kc = t >> 8, nt = (t >> 5) & 7, lane = t & 31;
  const int col = nt * 16 + (lane & 15), sel = lane >> 4;
  unsigned short* dst = P + t * 16;
#pragma unroll
  for (int i = 0; i < 16; ++i) {
    const int k = kc * 32 + 16 * sel + i;
    const float val = ((k >> 5) == (col >> 5))
        ? R[(k >> 5) * 1024 + (k & 31) * 32 + (col & 31)] : 0.0f;
    dst[i] = f2bf(val);
  }
}

// ---------------- WMMA GEMM: C[M x 128] = A[M x 128] @ Wpacked (+bias) ----------------
// mode 0: C = A@W + bias      mode 1: fused HGT update: gelu, skip-mix vs Hold, layernorm
__global__ __launch_bounds__(32) void k_gemm128(
    const float* __restrict__ A, const unsigned short* __restrict__ P,
    const float* __restrict__ bias, const float* Hold,
    const float* __restrict__ skipP, const float* __restrict__ lnS,
    const float* __restrict__ lnB, float* C, int mode) {
  const int tile = blockIdx.x;
  const int lane = threadIdx.x;
  const int r = lane & 15, sel = lane >> 4;
  const float* arow = A + (size_t)(tile * 16 + r) * NHID;
  v8f acc[8] = {};
#pragma unroll
  for (int kc = 0; kc < 4; ++kc) {
    BFrag af;
    const int k0 = kc * 32 + 16 * sel;
    const float4 f0 = *reinterpret_cast<const float4*>(arow + k0);
    const float4 f1 = *reinterpret_cast<const float4*>(arow + k0 + 4);
    const float4 f2 = *reinterpret_cast<const float4*>(arow + k0 + 8);
    const float4 f3 = *reinterpret_cast<const float4*>(arow + k0 + 12);
    af.us[0] = f2bf(f0.x);  af.us[1] = f2bf(f0.y);
    af.us[2] = f2bf(f0.z);  af.us[3] = f2bf(f0.w);
    af.us[4] = f2bf(f1.x);  af.us[5] = f2bf(f1.y);
    af.us[6] = f2bf(f1.z);  af.us[7] = f2bf(f1.w);
    af.us[8] = f2bf(f2.x);  af.us[9] = f2bf(f2.y);
    af.us[10] = f2bf(f2.z); af.us[11] = f2bf(f2.w);
    af.us[12] = f2bf(f3.x); af.us[13] = f2bf(f3.y);
    af.us[14] = f2bf(f3.z); af.us[15] = f2bf(f3.w);
#pragma unroll
    for (int nt = 0; nt < 8; ++nt) {
      BFrag bf;
      const uint4* pp = reinterpret_cast<const uint4*>(P + (((kc * 8 + nt) * 32 + lane) << 4));
      bf.q[0] = pp[0]; bf.q[1] = pp[1];
      acc[nt] = __builtin_amdgcn_wmma_f32_16x16x32_bf16(
          false, af.v, false, bf.v, (short)0, acc[nt], false, false);
    }
  }
  if (mode == 0) {
#pragma unroll
    for (int nt = 0; nt < 8; ++nt) {
      const int col = nt * 16 + r;
      const float b = bias ? bias[col] : 0.0f;
#pragma unroll
      for (int j = 0; j < 8; ++j) {
        const int row = tile * 16 + j + 8 * sel;
        C[(size_t)row * NHID + col] = acc[nt][j] + b;
      }
    }
  } else {
    const float alpha = 1.0f / (1.0f + expf(-skipP[0]));
#pragma unroll
    for (int nt = 0; nt < 8; ++nt) {
      const int col = nt * 16 + r;
      const float b = bias[col];
#pragma unroll
      for (int j = 0; j < 8; ++j) {
        const int row = tile * 16 + j + 8 * sel;
        const float t = acc[nt][j] + b;
        const float ho = Hold[(size_t)row * NHID + col];
        acc[nt][j] = alpha * gelu_exact(t) + (1.0f - alpha) * ho;
      }
    }
#pragma unroll
    for (int j = 0; j < 8; ++j) {
      float s1 = 0.f, s2 = 0.f;
#pragma unroll
      for (int nt = 0; nt < 8; ++nt) { const float x = acc[nt][j]; s1 += x; s2 += x * x; }
#pragma unroll
      for (int off = 1; off < 16; off <<= 1) {   // reduce across the 16-lane half
        s1 += __shfl_xor(s1, off, 32);
        s2 += __shfl_xor(s2, off, 32);
      }
      const float mu = s1 * (1.0f / 128.0f);
      const float var = s2 * (1.0f / 128.0f) - mu * mu;
      const float rstd = rsqrtf(var + 1e-5f);
      const int row = tile * 16 + j + 8 * sel;
#pragma unroll
      for (int nt = 0; nt < 8; ++nt) {
        const int col = nt * 16 + r;
        C[(size_t)row * NHID + col] = (acc[nt][j] - mu) * rstd * lnS[col] + lnB[col];
      }
    }
  }
}

// ---------------- init / pooling ----------------
__global__ void k_init_h1(const float* __restrict__ tok, float* __restrict__ h1) {
  int i = blockIdx.x * blockDim.x + threadIdx.x;
  if (i < NGENE * NHID) h1[i] = tok[i & 127];
}
__global__ void k_vec_copy(const float* __restrict__ s, float* __restrict__ d) {
  d[threadIdx.x] = s[threadIdx.x];
}
__global__ void k_wsi_pool(const float* __restrict__ patches,
                           const unsigned char* __restrict__ mask,
                           float* __restrict__ pooled) {
  const int d = blockIdx.x * blockDim.x + threadIdx.x;
  if (d >= WSIDIM) return;
  float s = 0.f, c = 0.f;
  for (int p = 0; p < NPATCH; ++p) {
    const float m = mask[p] ? 1.0f : 0.0f;
    s += m * patches[(size_t)p * WSIDIM + d];
    c += m;
  }
  pooled[d] = s / fmaxf(c, 1.0f);
}
__global__ void k_wsi_init(const float* __restrict__ tok2, const float* __restrict__ pooled,
                           const float* __restrict__ W, const float* __restrict__ b,
                           float* __restrict__ h2) {
  __shared__ float sp[WSIDIM];
  for (int i = threadIdx.x; i < WSIDIM; i += blockDim.x) sp[i] = pooled[i];
  __syncthreads();
  const int j = threadIdx.x;
  float a = b[j];
  for (int k = 0; k < WSIDIM; ++k) a += sp[k] * W[(size_t)k * NHID + j];
  h2[j] = tok2[j] + gelu_exact(a);
}
__global__ void k_wsum_gg(const int* __restrict__ dst, const float* __restrict__ w,
                          float* __restrict__ wsum, int n) {
  for (int e = blockIdx.x * blockDim.x + threadIdx.x; e < n; e += gridDim.x * blockDim.x)
    atomicAdd(&wsum[dst[e]], w[e]);
}
__global__ void k_wsum_gp(const float* __restrict__ w, float* __restrict__ out, int n) {
  const int g = blockIdx.x * blockDim.x + threadIdx.x;
  float v = (g < n) ? w[g] : 0.f;
#pragma unroll
  for (int off = 16; off >= 1; off >>= 1) v += __shfl_xor(v, off, 32);
  if ((threadIdx.x & 31) == 0) atomicAdd(out, v);
}

// ---------------- small node projections ----------------
__global__ void k_patient_pre(const float* __restrict__ h0, const float* __restrict__ Wq_p,
                              const float* __restrict__ bq_p, const float* __restrict__ relA,
                              float* __restrict__ qpat, float* __restrict__ rq) {
  __shared__ float sh[128];
  __shared__ float sq[128];
  const int j = threadIdx.x;
  sh[j] = h0[j];
  __syncthreads();
  float a = bq_p[j];
  for (int k = 0; k < 128; ++k) a += sh[k] * Wq_p[k * 128 + j];
  qpat[j] = a; sq[j] = a;
  __syncthreads();
  const int h = j >> 5, d = j & 31;
  const float* R = relA + h * 1024 + d * 32;   // rq = relA @ qpat (per head)
  float r = 0.f;
#pragma unroll
  for (int f = 0; f < 32; ++f) r += R[f] * sq[h * 32 + f];
  rq[j] = r;
}
__global__ void k_wsi_proj(const float* __restrict__ h2, const float* __restrict__ Wv_w,
                           const float* __restrict__ bv_w, const float* __restrict__ relM,
                           float* __restrict__ vtwp) {
  __shared__ float sh[128];
  __shared__ float sv[128];
  const int j = threadIdx.x;
  sh[j] = h2[j];
  __syncthreads();
  float a = bv_w[j];
  for (int k = 0; k < 128; ++k) a += sh[k] * Wv_w[k * 128 + j];
  sv[j] = a;
  __syncthreads();
  const int h = j >> 5, f = j & 31;
  float vt = 0.f;
#pragma unroll
  for (int d = 0; d < 32; ++d) vt += sv[h * 32 + d] * relM[h * 1024 + d * 32 + f];
  vtwp[j] = vt;
}

// ---------------- gene-gene edge attention (3-pass atomic segment softmax) ----------------
__global__ void k_gg_logits(const int* __restrict__ src, const int* __restrict__ dst,
                            const float* __restrict__ kt, const float* __restrict__ q,
                            const float* __restrict__ pri, float* __restrict__ logits,
                            unsigned* __restrict__ maxb, int n) {
  for (int idx = blockIdx.x * blockDim.x + threadIdx.x; idx < n; idx += gridDim.x * blockDim.x) {
    const int e = idx >> 2, h = idx & 3;
    const int s = src[e], d = dst[e];
    const float* kp = kt + (size_t)s * 128 + h * 32;
    const float* qp = q + (size_t)d * 128 + h * 32;
    float a = 0.f;
#pragma unroll
    for (int i = 0; i < 32; ++i) a += kp[i] * qp[i];
    const float lg = a * pri[h] * INV_SQRT_DK;
    logits[idx] = lg;
    atomicMax(&maxb[d * 4 + h], fkey(lg));
  }
}
__global__ void k_gg_expsum(const int* __restrict__ dst, const float* __restrict__ logits,
                            const unsigned* __restrict__ maxb, float* __restrict__ sumb, int n) {
  for (int idx = blockIdx.x * blockDim.x + threadIdx.x; idx < n; idx += gridDim.x * blockDim.x) {
    const int e = idx >> 2, h = idx & 3;
    const int d = dst[e];
    atomicAdd(&sumb[d * 4 + h], expf(logits[idx] - fdec(maxb[d * 4 + h])));
  }
}
__global__ void k_gg_scatter(const int* __restrict__ src, const int* __restrict__ dst,
                             const float* __restrict__ w, const float* __restrict__ wsum,
                             const float* __restrict__ logits, const unsigned* __restrict__ maxb,
                             const float* __restrict__ sumb, const float* __restrict__ vt,
                             float* __restrict__ msg, int n) {
  for (int idx = blockIdx.x * blockDim.x + threadIdx.x; idx < n; idx += gridDim.x * blockDim.x) {
    const int e = idx >> 2, h = idx & 3;
    const int s = src[e], d = dst[e];
    const float m = fdec(maxb[d * 4 + h]);
    const float p = expf(logits[idx] - m) / (sumb[d * 4 + h] + 1e-9f);
    const float c = p * (w[e] / (wsum[d] + 1e-9f));
    const float* vp = vt + (size_t)s * 128 + h * 32;
    float* mp = msg + (size_t)d * 128 + h * 32;
#pragma unroll
    for (int i = 0; i < 32; ++i) atomicAdd(&mp[i], c * vp[i]);
  }
}

// ---------------- gene -> patient edges (single segment) ----------------
__global__ void k_gp_logits(const int* __restrict__ gsrc, const float* __restrict__ kb,
                            const float* __restrict__ rq, const float* __restrict__ pri,
                            float* __restrict__ logits, unsigned* __restrict__ maxg, int n) {
  const int g = blockIdx.x * blockDim.x + threadIdx.x;
  const int lane = threadIdx.x & 31;
  const bool act = g < n;
  int s = 0;
  if (act) s = gsrc[g];
  float lg[4];
#pragma unroll
  for (int h = 0; h < 4; ++h) {
    float a = -INFINITY;
    if (act) {
      const float* kp = kb + (size_t)s * 128 + h * 32;
      const float* rp = rq + h * 32;
      float acc = 0.f;
#pragma unroll
      for (int i = 0; i < 32; ++i) acc += kp[i] * rp[i];
      a = acc * pri[h] * INV_SQRT_DK;
      logits[g * 4 + h] = a;
    }
    lg[h] = a;
  }
#pragma unroll
  for (int h = 0; h < 4; ++h) {
    float m = lg[h];
#pragma unroll
    for (int off = 16; off >= 1; off >>= 1) m = fmaxf(m, __shfl_xor(m, off, 32));
    if (lane == 0) atomicMax(&maxg[h], fkey(m));
  }
}
__global__ void k_gp_expsum(const float* __restrict__ logits, const unsigned* __restrict__ maxg,
                            float* __restrict__ sumg, int n) {
  const int g = blockIdx.x * blockDim.x + threadIdx.x;
  const int lane = threadIdx.x & 31;
  const bool act = g < n;
#pragma unroll
  for (int h = 0; h < 4; ++h) {
    float e = act ? expf(logits[g * 4 + h] - fdec(maxg[h])) : 0.f;
#pragma unroll
    for (int off = 16; off >= 1; off >>= 1) e += __shfl_xor(e, off, 32);
    if (lane == 0) atomicAdd(&sumg[h], e);
  }
}
// u = sum_g (p*wn) * v_g  (rel_msg applied later in patient_update: (sum c v) @ R)
__global__ void k_gp_scatter(const int* __restrict__ gsrc, const float* __restrict__ gw,
                             const float* __restrict__ wsgp, const float* __restrict__ logits,
                             const unsigned* __restrict__ maxg, const float* __restrict__ sumg,
                             const float* __restrict__ v, float* __restrict__ aggu, int n) {
  __shared__ float sAgg[128];
  if (threadIdx.x < 128) sAgg[threadIdx.x] = 0.f;
  __syncthreads();
  const float ws = wsgp[0];
  for (int g = blockIdx.x * blockDim.x + threadIdx.x; g < n; g += gridDim.x * blockDim.x) {
    const int s = gsrc[g];
    const float wn = gw[g] / (ws + 1e-9f);
#pragma unroll
    for (int h = 0; h < 4; ++h) {
      const float p = expf(logits[g * 4 + h] - fdec(maxg[h])) / (sumg[h] + 1e-9f);
      const float c = p * wn;
      const float* vp = v + (size_t)s * 128 + h * 32;
#pragma unroll
      for (int d = 0; d < 32; ++d) atomicAdd(&sAgg[h * 32 + d], c * vp[d]);
    }
  }
  __syncthreads();
  if (threadIdx.x < 128) atomicAdd(&aggu[threadIdx.x], sAgg[threadIdx.x]);
}

// ---------------- small node updates ----------------
__global__ void k_patient_update(const float* __restrict__ aggu, const float* __restrict__ relM,
                                 const float* __restrict__ vtwp, const float* __restrict__ wpw,
                                 const float* __restrict__ Wa_p, const float* __restrict__ ba_p,
                                 const float* __restrict__ skp, const float* __restrict__ lns,
                                 const float* __restrict__ lnb, float* h0) {
  __shared__ float st[128];
  __shared__ float sx[128];
  __shared__ float red[2];
  const int j = threadIdx.x;
  const int h = j >> 5, f = j & 31;
  float tg = 0.f;
#pragma unroll
  for (int d = 0; d < 32; ++d) tg += aggu[h * 32 + d] * relM[h * 1024 + d * 32 + f];
  const float w0 = wpw[0];
  const float wn = w0 / (w0 + 1e-9f);                  // wp softmax p==1 (single edge)
  const float t = 0.5f * (tg + wn * vtwp[j]);          // mean over {gp, wp}
  st[j] = t;
  __syncthreads();
  float a = ba_p[j];
  for (int k = 0; k < 128; ++k) a += st[k] * Wa_p[k * 128 + j];
  const float alpha = 1.0f / (1.0f + expf(-skp[0]));
  const float x = alpha * gelu_exact(a) + (1.0f - alpha) * h0[j];
  sx[j] = x;
  __syncthreads();
  if (j == 0) {
    float s1 = 0.f, s2 = 0.f;
    for (int k = 0; k < 128; ++k) { s1 += sx[k]; s2 += sx[k] * sx[k]; }
    const float mu = s1 / 128.f;
    red[0] = mu; red[1] = s2 / 128.f - mu * mu;
  }
  __syncthreads();
  h0[j] = (x - red[0]) * rsqrtf(red[1] + 1e-5f) * lns[j] + lnb[j];
}
__global__ void k_wsi_update(const float* __restrict__ ba_w, const float* __restrict__ skp,
                             const float* __restrict__ lns, const float* __restrict__ lnb,
                             float* h2) {
  __shared__ float sx[128];
  __shared__ float red[2];
  const int j = threadIdx.x;
  const float alpha = 1.0f / (1.0f + expf(-skp[0]));
  const float x = alpha * gelu_exact(ba_w[j]) + (1.0f - alpha) * h2[j];  // t == 0 (no in-edges)
  sx[j] = x;
  __syncthreads();
  if (j == 0) {
    float s1 = 0.f, s2 = 0.f;
    for (int k = 0; k < 128; ++k) { s1 += sx[k]; s2 += sx[k] * sx[k]; }
    const float mu = s1 / 128.f;
    red[0] = mu; red[1] = s2 / 128.f - mu * mu;
  }
  __syncthreads();
  h2[j] = (x - red[0]) * rsqrtf(red[1] + 1e-5f) * lns[j] + lnb[j];
}
__global__ void k_output(const float* __restrict__ h0, const float* __restrict__ W,
                         const float* __restrict__ b, float* __restrict__ out) {
  __shared__ float sh[128];
  const int j = threadIdx.x;
  sh[j] = h0[j];
  __syncthreads();
  if (j < 4) {
    float a = b[j];
    for (int k = 0; k < 128; ++k) a += sh[k] * W[k * 4 + j];
    out[j] = a;
  }
}

extern "C" void kernel_launch(void* const* d_in, const int* in_sizes, int n_in,
                              void* d_out, int out_size, void* d_ws, size_t ws_size,
                              hipStream_t stream) {
  (void)in_sizes; (void)n_in; (void)out_size; (void)ws_size;
  const int* gg_src = (const int*)d_in[0];
  const int* gg_dst = (const int*)d_in[1];
  const float* gg_w = (const float*)d_in[2];
  const int* gp_src = (const int*)d_in[3];
  const float* gp_w = (const float*)d_in[5];
  const float* wp_w = (const float*)d_in[8];
  const float* wsi_patches = (const float*)d_in[9];
  const unsigned char* wsi_mask = (const unsigned char*)d_in[10];
  const float* type_tokens = (const float*)d_in[11];
  const float* wsi_W = (const float*)d_in[12];
  const float* wsi_b = (const float*)d_in[13];
  const float* Wk = (const float*)d_in[14];
  const float* Wq = (const float*)d_in[15];
  const float* Wv = (const float*)d_in[16];
  const float* Wa = (const float*)d_in[17];
  const float* bk = (const float*)d_in[18];
  const float* bq = (const float*)d_in[19];
  const float* bv = (const float*)d_in[20];
  const float* ba = (const float*)d_in[21];
  const float* rel_pri = (const float*)d_in[22];
  const float* rel_att = (const float*)d_in[23];
  const float* rel_msg = (const float*)d_in[24];
  const float* skip = (const float*)d_in[25];
  const float* ln_scale = (const float*)d_in[26];
  const float* ln_bias = (const float*)d_in[27];
  const float* out_W = (const float*)d_in[28];
  const float* out_b = (const float*)d_in[29];
  float* out = (float*)d_out;

  char* ws = (char*)d_ws;
  size_t off = 0;
  auto alloc = [&](size_t bytes) -> char* {
    char* p = ws + off;
    off = (off + bytes + 255) & ~(size_t)255;
    return p;
  };
  const size_t NODE = (size_t)NGENE * NHID * sizeof(float);
  float* h1 = (float*)alloc(NODE);
  float* kb = (float*)alloc(NODE);
  float* qb = (float*)alloc(NODE);
  float* vb = (float*)alloc(NODE);
  float* ktgg = (float*)alloc(NODE);
  float* vtgg = (float*)alloc(NODE);
  float* msg = (float*)alloc(NODE);
  float* lg_gg = (float*)alloc((size_t)EGG * 4 * sizeof(float));
  unsigned* maxb = (unsigned*)alloc((size_t)NGENE * 4 * sizeof(unsigned));
  float* sumb = (float*)alloc((size_t)NGENE * 4 * sizeof(float));
  float* wsum = (float*)alloc((size_t)NGENE * sizeof(float));
  float* lg_gp = (float*)alloc((size_t)NGENE * 4 * sizeof(float));
  unsigned short* pk = (unsigned short*)alloc(16384 * 2);
  unsigned short* pq = (unsigned short*)alloc(16384 * 2);
  unsigned short* pv = (unsigned short*)alloc(16384 * 2);
  unsigned short* pa = (unsigned short*)alloc(16384 * 2);
  unsigned short* pra = (unsigned short*)alloc(16384 * 2);
  unsigned short* prm = (unsigned short*)alloc(16384 * 2);
  float* pooled = (float*)alloc(WSIDIM * sizeof(float));
  float* h0 = (float*)alloc(512);
  float* h2 = (float*)alloc(512);
  float* qpat = (float*)alloc(512);
  float* rqv = (float*)alloc(512);
  float* vtwp = (float*)alloc(512);
  float* aggu = (float*)alloc(512);
  unsigned* maxg = (unsigned*)alloc(64);
  float* sumg = (float*)alloc(64);
  float* wsgp = (float*)alloc(64);

  // init + pooling + dst-weight sums (layer-invariant)
  hipMemsetAsync(wsum, 0, NGENE * sizeof(float), stream);
  hipMemsetAsync(wsgp, 0, sizeof(float), stream);
  k_init_h1<<<(NGENE * NHID + 255) / 256, 256, 0, stream>>>(type_tokens + 128, h1);
  k_vec_copy<<<1, 128, 0, stream>>>(type_tokens, h0);
  k_wsi_pool<<<(WSIDIM + 255) / 256, 256, 0, stream>>>(wsi_patches, wsi_mask, pooled);
  k_wsi_init<<<1, 128, 0, stream>>>(type_tokens + 256, pooled, wsi_W, wsi_b, h2);
  k_wsum_gg<<<(EGG + 255) / 256, 256, 0, stream>>>(gg_dst, gg_w, wsum, EGG);
  k_wsum_gp<<<(NGENE + 255) / 256, 256, 0, stream>>>(gp_w, wsgp, NGENE);

  const int MT = NGENE / 16;           // 3125 tiles, one wave each
  const int NEH = EGG * 4;
  const int GEH = (NEH + 255) / 256;
  const int GG = (NGENE + 255) / 256;

  for (int l = 0; l < 2; ++l) {
    const size_t gM = (size_t)(l * 3 + 1) * 16384;   // gene weight block
    // swizzle weights into WMMA fragment order
    k_pack_w<<<4, 256, 0, stream>>>(Wk + gM, pk);
    k_pack_w<<<4, 256, 0, stream>>>(Wq + gM, pq);
    k_pack_w<<<4, 256, 0, stream>>>(Wv + gM, pv);
    k_pack_w<<<4, 256, 0, stream>>>(Wa + gM, pa);
    k_pack_bd<<<4, 256, 0, stream>>>(rel_att + (size_t)(l * 3 + 0) * 4096, pra);
    k_pack_bd<<<4, 256, 0, stream>>>(rel_msg + (size_t)(l * 3 + 0) * 4096, prm);
    // gene projections + hoisted relation maps (WMMA)
    k_gemm128<<<MT, 32, 0, stream>>>(h1, pk, bk + (l * 3 + 1) * 128, nullptr, nullptr, nullptr, nullptr, kb, 0);
    k_gemm128<<<MT, 32, 0, stream>>>(h1, pq, bq + (l * 3 + 1) * 128, nullptr, nullptr, nullptr, nullptr, qb, 0);
    k_gemm128<<<MT, 32, 0, stream>>>(h1, pv, bv + (l * 3 + 1) * 128, nullptr, nullptr, nullptr, nullptr, vb, 0);
    k_gemm128<<<MT, 32, 0, stream>>>(kb, pra, nullptr, nullptr, nullptr, nullptr, nullptr, ktgg, 0);
    k_gemm128<<<MT, 32, 0, stream>>>(vb, prm, nullptr, nullptr, nullptr, nullptr, nullptr, vtgg, 0);
    // small node projections
    k_patient_pre<<<1, 128, 0, stream>>>(h0, Wq + (size_t)(l * 3 + 0) * 16384,
                                         bq + (l * 3 + 0) * 128,
                                         rel_att + (size_t)(l * 3 + 1) * 4096, qpat, rqv);
    k_wsi_proj<<<1, 128, 0, stream>>>(h2, Wv + (size_t)(l * 3 + 2) * 16384,
                                      bv + (l * 3 + 2) * 128,
                                      rel_msg + (size_t)(l * 3 + 2) * 4096, vtwp);
    // clear per-layer accumulators
    hipMemsetAsync(maxb, 0, (size_t)NGENE * 4 * sizeof(unsigned), stream);
    hipMemsetAsync(sumb, 0, (size_t)NGENE * 4 * sizeof(float), stream);
    hipMemsetAsync(msg, 0, NODE, stream);
    hipMemsetAsync(maxg, 0, 4 * sizeof(unsigned), stream);
    hipMemsetAsync(sumg, 0, 4 * sizeof(float), stream);
    hipMemsetAsync(aggu, 0, 128 * sizeof(float), stream);
    // gene-gene attention
    k_gg_logits<<<GEH, 256, 0, stream>>>(gg_src, gg_dst, ktgg, qb,
                                         rel_pri + (l * 3 + 0) * 4, lg_gg, maxb, NEH);
    k_gg_expsum<<<GEH, 256, 0, stream>>>(gg_dst, lg_gg, maxb, sumb, NEH);
    k_gg_scatter<<<GEH, 256, 0, stream>>>(gg_src, gg_dst, gg_w, wsum, lg_gg, maxb, sumb,
                                          vtgg, msg, NEH);
    // gene->patient attention
    k_gp_logits<<<GG, 256, 0, stream>>>(gp_src, kb, rqv, rel_pri + (l * 3 + 1) * 4,
                                        lg_gp, maxg, NGENE);
    k_gp_expsum<<<GG, 256, 0, stream>>>(lg_gp, maxg, sumg, NGENE);
    k_gp_scatter<<<64, 256, 0, stream>>>(gp_src, gp_w, wsgp, lg_gp, maxg, sumg, vb, aggu, NGENE);
    // node updates
    k_patient_update<<<1, 128, 0, stream>>>(aggu, rel_msg + (size_t)(l * 3 + 1) * 4096, vtwp,
                                            wp_w, Wa + (size_t)(l * 3 + 0) * 16384,
                                            ba + (l * 3 + 0) * 128, skip + (l * 3 + 0),
                                            ln_scale + (l * 3 + 0) * 128,
                                            ln_bias + (l * 3 + 0) * 128, h0);
    k_wsi_update<<<1, 128, 0, stream>>>(ba + (l * 3 + 2) * 128, skip + (l * 3 + 2),
                                        ln_scale + (l * 3 + 2) * 128,
                                        ln_bias + (l * 3 + 2) * 128, h2);
    // fused gene update: gelu(msg@Wa+ba), skip-mix with h1, layernorm -> h1 (in place)
    k_gemm128<<<MT, 32, 0, stream>>>(msg, pa, ba + (l * 3 + 1) * 128, h1, skip + (l * 3 + 1),
                                     ln_scale + (l * 3 + 1) * 128,
                                     ln_bias + (l * 3 + 1) * 128, h1, 1);
  }
  k_output<<<1, 128, 0, stream>>>(h0, out_W, out_b, out);
}
